// NTXentLoss_63032940036243
// MI455X (gfx1250) — compile-verified, matrix operand order checked by reading, not run on hardware
//
#include <hip/hip_runtime.h>
#include <hip/hip_bf16.h>

// ---------------------------------------------------------------------------
// NT-Xent loss, fused flash-softmax over sim = (zn @ zn^T)/T, CDNA5 WMMA.
// ---------------------------------------------------------------------------

typedef __attribute__((ext_vector_type(16))) _Float16 v16h;
typedef __attribute__((ext_vector_type(8)))  _Float16 v8h;
typedef __attribute__((ext_vector_type(8)))  float    v8f;
typedef __attribute__((ext_vector_type(4)))  int      v4i;

#define AS1 __attribute__((address_space(1)))
#define AS3 __attribute__((address_space(3)))

#define DIM        256          // feature dim (fixed by reference)
#define KCH        8            // 256 / 32 K-chunks per 16x16x32 WMMA
#define LROW       264          // LDS row stride in halfs (256 + 8 pad -> no bank conflicts)
#define ROWS_PER_WG 128         // 8 waves x 16 rows
#define COLSPLIT   2            // column halves per row block (parallelism)
#define INV_TEMP   10.0f        // 1 / 0.1
#define NEG_BIG    -9.0e15f

#if __has_builtin(__builtin_amdgcn_global_load_async_to_lds_b128)
#define USE_ASYNC_LDS 1
#else
#define USE_ASYNC_LDS 0
#endif

#if __has_builtin(__builtin_amdgcn_sched_barrier)
#define SCHED_FENCE() __builtin_amdgcn_sched_barrier(0)
#else
#define SCHED_FENCE()
#endif

// ---------------------------------------------------------------------------
// 1) zero the global accumulators
__global__ void ntx_init_kernel(float* acc) {
    if (threadIdx.x < 2) acc[threadIdx.x] = 0.0f;
}

// ---------------------------------------------------------------------------
// 2) row-normalize z = concat(z_i, z_j) -> f16 zh [NN x 256]
__global__ void __launch_bounds__(256)
ntx_normalize_kernel(const float* __restrict__ zi, const float* __restrict__ zj,
                     _Float16* __restrict__ zh, int B) {
    const int r = blockIdx.x;
    const int t = threadIdx.x;                       // 256 threads = 256 elems
    const float* src = (r < B) ? (zi + (size_t)r * DIM)
                               : (zj + (size_t)(r - B) * DIM);
    float v  = src[t];
    float sq = v * v;
    #pragma unroll
    for (int d = 16; d > 0; d >>= 1) sq += __shfl_down(sq, d);
    __shared__ float wsum[8];
    __shared__ float ninv;
    if ((t & 31) == 0) wsum[t >> 5] = sq;
    __syncthreads();
    if (t == 0) {
        float ss = 0.f;
        #pragma unroll
        for (int i = 0; i < 8; ++i) ss += wsum[i];
        ninv = 1.0f / fmaxf(sqrtf(ss), 1e-8f);
    }
    __syncthreads();
    zh[(size_t)r * DIM + t] = (_Float16)(v * ninv);
}

// ---------------------------------------------------------------------------
// 3) fused WMMA GEMM + online softmax stats.
//    blockIdx.x = rowblk * COLSPLIT + half.
//    Each wave owns 16 rows; WG stages one 16-column tile of zn in LDS
//    (double buffered, async global->LDS) and all 8 waves consume it.
__global__ void __launch_bounds__(256)
ntx_sim_softmax_kernel(const _Float16* __restrict__ zh,
                       float* __restrict__ mpart, float* __restrict__ spart,
                       float* __restrict__ ppart, int B, int NN) {
    __shared__ __align__(16) _Float16 smem[2][16][LROW];

    const int tid  = threadIdx.x;
    const int lane = tid & 31;
    const int wave = tid >> 5;
    const int rowblk = blockIdx.x / COLSPLIT;
    const int half   = blockIdx.x % COLSPLIT;
    const int m0     = rowblk * ROWS_PER_WG + wave * 16;

    const int jrow = lane & 15;      // A: row in tile, B: column in tile
    const int hi   = lane >> 4;      // half-wave select
    const int kb   = hi * 8;         // per-lane K sub-offset (ISA 16-bit layout)

    // ---- preload A fragments (16 rows, full K) into registers -------------
    v16h afrag[KCH];
    {
        const _Float16* arow = zh + (size_t)(m0 + jrow) * DIM;
        #pragma unroll
        for (int kc = 0; kc < KCH; ++kc) {
            v8h lo = *(const v8h*)(arow + kc * 32 + kb);
            v8h hh = *(const v8h*)(arow + kc * 32 + kb + 16);
            afrag[kc] = __builtin_shufflevector(lo, hh,
                0,1,2,3,4,5,6,7,8,9,10,11,12,13,14,15);
        }
    }

    float runmax[8], runsum[8], posv[8];
    #pragma unroll
    for (int v = 0; v < 8; ++v) {
        runmax[v] = -3.0e38f; runsum[v] = 0.0f; posv[v] = NEG_BIG;
    }

    const int tiles = NN / (16 * COLSPLIT);   // column tiles in this half
    const int t0    = half * tiles;

    // staging map: 256 threads x 16 halfs = 16 rows x 256 halfs per tile
    const int sj  = tid >> 4;
    const int ssg = (tid & 15) * 16;

    auto stage = [&](int b, int t) {
        const int n0 = (t0 + t) * 16;
        const _Float16* src = zh + (size_t)(n0 + sj) * DIM + ssg;
        _Float16* dst = &smem[b][sj][ssg];
#if USE_ASYNC_LDS
        // INST_OFFSET applies to both global and LDS addresses (ISA 08 §4.4)
        __builtin_amdgcn_global_load_async_to_lds_b128(
            (AS1 v4i*)(src), (AS3 v4i*)(dst), 0, 0);
        __builtin_amdgcn_global_load_async_to_lds_b128(
            (AS1 v4i*)(src), (AS3 v4i*)(dst), 16, 0);
#else
        v8h x0 = *(const v8h*)(src);
        v8h x1 = *(const v8h*)(src + 8);
        *(v8h*)(dst)     = x0;
        *(v8h*)(dst + 8) = x1;
#endif
    };

    auto wait_stage = [&]() {
#if USE_ASYNC_LDS
#if __has_builtin(__builtin_amdgcn_s_wait_asynccnt)
        __builtin_amdgcn_s_wait_asynccnt(0);
#else
        asm volatile("s_wait_asynccnt 0" ::: "memory");
#endif
#endif
    };

    stage(0, 0);
    wait_stage();
    __syncthreads();

    const int rowbase = m0 + kb;              // lane's rows: rowbase + v

    for (int t = 0; t < tiles; ++t) {
        const int b = t & 1;
        if (t + 1 < tiles) stage(b ^ 1, t + 1);

        // ---- load ALL B fragments (one ds clause, one dscnt wait) --------
        v16h bfrag[KCH];
        const _Float16* brow = &smem[b][jrow][kb];
        #pragma unroll
        for (int kc = 0; kc < KCH; ++kc) {
            v8h blo = *(const v8h*)(brow + kc * 32);
            v8h bhi = *(const v8h*)(brow + kc * 32 + 16);
            bfrag[kc] = __builtin_shufflevector(blo, bhi,
                0,1,2,3,4,5,6,7,8,9,10,11,12,13,14,15);
        }
        // Pin the schedule: loads must stay above, WMMAs below. This forces
        // all 8 fragments live -> one s_wait_dscnt, 8 back-to-back WMMAs.
        SCHED_FENCE();

        // Two independent accumulation chains so the XDL pipe can overlap
        // them instead of stalling on the D->C dependence each cycle.
        v8f c0 = {}, c1 = {};
        #pragma unroll
        for (int kc = 0; kc < KCH; kc += 2) {
            c0 = __builtin_amdgcn_wmma_f32_16x16x32_f16(
                    false, afrag[kc],     false, bfrag[kc],     (short)0, c0, false, false);
            c1 = __builtin_amdgcn_wmma_f32_16x16x32_f16(
                    false, afrag[kc + 1], false, bfrag[kc + 1], (short)0, c1, false, false);
        }
        SCHED_FENCE();
        v8f c = c0 + c1;

        const int col = (t0 + t) * 16 + jrow;
        #pragma unroll
        for (int v = 0; v < 8; ++v) {
            const int row = rowbase + v;
            const float val = c[v] * INV_TEMP;
            const bool isdiag = (col == row);
            const bool ispos  = (col == row + B) || (row == col + B);
            if (!isdiag) {
                if (ispos) posv[v] = val;
                if (val > runmax[v]) {
                    runsum[v] *= __expf(runmax[v] - val);
                    runmax[v]  = val;
                }
                runsum[v] += __expf(val - runmax[v]);
            }
        }
        wait_stage();
        __syncthreads();
    }

    // ---- merge softmax states across the 16 column-lanes ------------------
    #pragma unroll
    for (int v = 0; v < 8; ++v) {
        float m = runmax[v], s = runsum[v], p = posv[v];
        #pragma unroll
        for (int dlt = 1; dlt < 16; dlt <<= 1) {
            float om = __shfl_xor(m, dlt);
            float os = __shfl_xor(s, dlt);
            float op = __shfl_xor(p, dlt);
            float nm = fmaxf(m, om);
            s = s * __expf(m - nm) + os * __expf(om - nm);
            m = nm;
            p = fmaxf(p, op);
        }
        if (jrow == 0) {
            const size_t o = (size_t)half * NN + (rowbase + v);
            mpart[o] = m; spart[o] = s; ppart[o] = p;
        }
    }
}

// ---------------------------------------------------------------------------
// 4) merge the COLSPLIT partial states, compute masked loss, reduce
__global__ void __launch_bounds__(256)
ntx_merge_kernel(const float* __restrict__ mpart, const float* __restrict__ spart,
                 const float* __restrict__ ppart, const unsigned char* __restrict__ mask,
                 float* __restrict__ acc, int B, int NN) {
    const int r = blockIdx.x * blockDim.x + threadIdx.x;
    float lossv = 0.0f, cntv = 0.0f;
    if (r < NN) {
        const float m0 = mpart[r], m1 = mpart[NN + r];
        const float s0 = spart[r], s1 = spart[NN + r];
        const float p  = fmaxf(ppart[r], ppart[NN + r]);
        const float m  = fmaxf(m0, m1);
        const float s  = s0 * __expf(m0 - m) + s1 * __expf(m1 - m);
        const float loss = -(p - m - __logf(s));
        const float mk = mask[r % B] ? 1.0f : 0.0f;   // NN = 2B
        lossv = loss * mk;
        cntv  = mk;
    }
    #pragma unroll
    for (int d = 16; d > 0; d >>= 1) {
        lossv += __shfl_down(lossv, d);
        cntv  += __shfl_down(cntv, d);
    }
    if ((threadIdx.x & 31) == 0) {
        atomicAdd(&acc[0], lossv);
        atomicAdd(&acc[1], cntv);
    }
}

// ---------------------------------------------------------------------------
// 5) finalize: mean over masked rows
__global__ void ntx_finalize_kernel(const float* acc, float* out) {
    const float total = acc[0], cnt = acc[1];
    out[0] = (cnt > 0.0f) ? (total / fmaxf(cnt, 1.0f)) : 0.0f;
}

// ---------------------------------------------------------------------------
extern "C" void kernel_launch(void* const* d_in, const int* in_sizes, int n_in,
                              void* d_out, int out_size, void* d_ws, size_t ws_size,
                              hipStream_t stream) {
    const float*         zi   = (const float*)d_in[0];
    const float*         zj   = (const float*)d_in[1];
    const unsigned char* mask = (const unsigned char*)d_in[2];
    float*               out  = (float*)d_out;

    const int B  = in_sizes[0] / DIM;   // 4096
    const int NN = 2 * B;               // 8192

    // workspace layout
    char* ws = (char*)d_ws;
    _Float16* zh = (_Float16*)ws;                       // NN*256 halfs = 4 MB
    size_t off = (size_t)NN * DIM * sizeof(_Float16);
    float* mpart = (float*)(ws + off);                  // COLSPLIT * NN
    float* spart = mpart + (size_t)COLSPLIT * NN;
    float* ppart = spart + (size_t)COLSPLIT * NN;
    float* acc   = ppart + (size_t)COLSPLIT * NN;       // 2 floats

    ntx_init_kernel<<<1, 32, 0, stream>>>(acc);
    ntx_normalize_kernel<<<NN, 256, 0, stream>>>(zi, zj, zh, B);
    ntx_sim_softmax_kernel<<<(NN / ROWS_PER_WG) * COLSPLIT, 256, 0, stream>>>(
        zh, mpart, spart, ppart, B, NN);
    ntx_merge_kernel<<<(NN + 255) / 256, 256, 0, stream>>>(
        mpart, spart, ppart, mask, acc, B, NN);
    ntx_finalize_kernel<<<1, 1, 0, stream>>>(acc, out);
}
